// minLSTM_15547781611893
// MI455X (gfx1250) — compile-verified, matrix operand order checked by reading, not run on hardware
//
#include <hip/hip_runtime.h>
#include <hip/hip_bf16.h>

// ---------------------------------------------------------------------------
// Problem constants (match reference)
// ---------------------------------------------------------------------------
#define DIMV    1024
#define INNERV  2048
#define SLEN    4096
#define BATCH   4
#define GSZ     16
#define KSZ     4
#define EPSV    1e-8f
#define NCHUNK  8
#define CHUNK   (SLEN / NCHUNK)   // 512

typedef __attribute__((ext_vector_type(16))) __bf16        v16bf;
typedef __attribute__((ext_vector_type(8)))  float         v8f;
typedef __attribute__((ext_vector_type(4))) unsigned int   u32x4;
typedef __attribute__((ext_vector_type(2))) unsigned int   u32x2;
typedef __attribute__((ext_vector_type(4)))  float         f32x4;

// round-to-nearest-even fp32 -> bf16
__device__ __forceinline__ unsigned short f2bf(float f) {
    unsigned u = __builtin_bit_cast(unsigned, f);
    u += 0x7FFFu + ((u >> 16) & 1u);
    return (unsigned short)(u >> 16);
}
__device__ __forceinline__ unsigned pack2(float a, float b) {
    return (unsigned)f2bf(a) | ((unsigned)f2bf(b) << 16);
}
__device__ __forceinline__ float softplus_f(float x) {
    return fmaxf(x, 0.f) + log1pf(__expf(-fabsf(x)));
}
__device__ __forceinline__ float silu_f(float x) {
    return x / (1.f + __expf(-x));
}

// ---------------------------------------------------------------------------
// bf16 WMMA GEMM:  C[M,N] = A[M,K] * B[N,K]^T + bias[N]
// MODE 0: A read directly (fp32 -> bf16 at staging)
// MODE 1: A elementwise-fused:  a = (hlast[b,c] + skip[c]*act[m,c]) * silu(z[m,c])
// Block = 256 threads = 8 waves (wave32). Block tile 128x128, wave tile 32x64.
// Double-buffered LDS, loads issued one FULL iteration ahead of consumption
// so no s_wait_loadcnt 0 lands next to the issue; one barrier per K-step.
// ---------------------------------------------------------------------------
template <int MODE>
__global__ __launch_bounds__(256)
void gemm_bf16_kernel(const float* __restrict__ A,
                      const float* __restrict__ Bw,
                      const float* __restrict__ bias,
                      float* __restrict__ C,
                      int M, int N, int K,
                      const float* __restrict__ zsrc,
                      const float* __restrict__ skip,
                      const float* __restrict__ hlast)
{
    // LDS tiles, 128 rows x 32 k (bf16), padded to 40 shorts/row (80B):
    // 16B-aligned fragment loads, conflict-free (20*L mod 64 distinct, L<16).
    __shared__ __align__(16) unsigned short lA[2][128 * 40];
    __shared__ __align__(16) unsigned short lB[2][128 * 40];

    const int tid   = threadIdx.x;
    const int lane  = tid & 31;
    const int wv    = tid >> 5;
    const int wm    = wv >> 1;            // 0..3  (rows of 32)
    const int wn    = wv & 1;             // 0..1  (cols of 64)
    const int tileM = blockIdx.y * 128;
    const int tileN = blockIdx.x * 128;
    const int half  = (lane >> 4) & 1;    // K-half selector per WMMA layout
    const int rl    = lane & 15;

    // per-thread staging coordinates (4 float4 chunks per thread)
    int srow[4], skc[4];
#pragma unroll
    for (int it = 0; it < 4; ++it) {
        const int idx = it * 256 + tid;
        srow[it] = idx >> 3;               // 0..127
        skc[it]  = (idx & 7) << 2;         // 0,4,...,28
    }

    v8f acc[2][4];
    const v8f vzero = {0.f, 0.f, 0.f, 0.f, 0.f, 0.f, 0.f, 0.f};
#pragma unroll
    for (int i = 0; i < 2; ++i)
#pragma unroll
        for (int j = 0; j < 4; ++j) acc[i][j] = vzero;

    f32x4 ra[4], rb[4], rz[4], rs[4], rh[4];

    // ---- issue global loads for K-tile k0 (consumed next iteration)
    auto loadTiles = [&](int k0) {
#pragma unroll
        for (int it = 0; it < 4; ++it) {
            const int gm = tileM + srow[it];
            const int ck = k0 + skc[it];
            const float* apt = A + (size_t)gm * K + ck;
            ra[it] = *(const f32x4*)apt;
            if (k0 + 32 < K) __builtin_prefetch(apt + 32, 0, 1);
            if constexpr (MODE == 1) {
                const int b = gm >> 12;    // row / 4096 (S = 4096)
                rz[it] = *(const f32x4*)(zsrc  + (size_t)gm * (2 * INNERV) + INNERV + ck);
                rs[it] = *(const f32x4*)(skip  + ck);
                rh[it] = *(const f32x4*)(hlast + (size_t)b * INNERV + ck);
            }
            const int gn = tileN + srow[it];
            rb[it] = *(const f32x4*)(Bw + (size_t)gn * K + ck);
        }
    };

    // ---- transform (MODE1), convert fp32->bf16, store to LDS buffer `buf`
    auto stashTiles = [&](int buf) {
#pragma unroll
        for (int it = 0; it < 4; ++it) {
            f32x4 v = ra[it];
            if constexpr (MODE == 1) {
#pragma unroll
                for (int e = 0; e < 4; ++e)
                    v[e] = (rh[it][e] + rs[it][e] * v[e]) * silu_f(rz[it][e]);
            }
            u32x2 p;
            p[0] = pack2(v[0], v[1]);
            p[1] = pack2(v[2], v[3]);
            *(u32x2*)&lA[buf][srow[it] * 40 + skc[it]] = p;

            u32x2 q;
            q[0] = pack2(rb[it][0], rb[it][1]);
            q[1] = pack2(rb[it][2], rb[it][3]);
            *(u32x2*)&lB[buf][srow[it] * 40 + skc[it]] = q;
        }
    };

    union Frag { u32x4 u[2]; v16bf v; };

    // ---- fragment gathers + 8 WMMAs from LDS buffer `buf`
    auto computeTile = [&](int buf) {
        Frag aF[2], bF[4];
        const unsigned short* pA = &lA[buf][0];
        const unsigned short* pB = &lB[buf][0];
#pragma unroll
        for (int i = 0; i < 2; ++i) {
            const int r = wm * 32 + i * 16 + rl;
            aF[i].u[0] = *(const u32x4*)&pA[r * 40 + half * 8];
            aF[i].u[1] = *(const u32x4*)&pA[r * 40 + half * 8 + 16];
        }
#pragma unroll
        for (int j = 0; j < 4; ++j) {
            const int cn = wn * 64 + j * 16 + rl;
            bF[j].u[0] = *(const u32x4*)&pB[cn * 40 + half * 8];
            bF[j].u[1] = *(const u32x4*)&pB[cn * 40 + half * 8 + 16];
        }
#pragma unroll
        for (int j = 0; j < 4; ++j)
#pragma unroll
            for (int i = 0; i < 2; ++i)
                acc[i][j] = __builtin_amdgcn_wmma_f32_16x16x32_bf16(
                    false, aF[i].v, false, bF[j].v, (short)0, acc[i][j],
                    false, false);
    };

    // ---- software pipeline over K (load -> [stash, sync, load next, compute])
    loadTiles(0);
    int cur = 0;
    for (int k0 = 0; k0 < K; k0 += 32) {
        stashTiles(cur);          // consumes loads from previous iteration
        __syncthreads();
        if (k0 + 32 < K)
            loadTiles(k0 + 32);   // in flight across the whole WMMA stage
        computeTile(cur);
        cur ^= 1;
    }

    // ---- store D tiles: VGPR d -> row (half?8:0)+d, col = lane&15
#pragma unroll
    for (int j = 0; j < 4; ++j) {
        const int n  = tileN + wn * 64 + j * 16 + rl;
        const float bn = bias[n];
#pragma unroll
        for (int i = 0; i < 2; ++i) {
            const int r0 = tileM + wm * 32 + i * 16 + half * 8;
#pragma unroll
            for (int d = 0; d < 8; ++d)
                C[(size_t)(r0 + d) * N + n] = acc[i][j][d] + bn;
        }
    }
}

// ---------------------------------------------------------------------------
// Pass 1: causal depthwise conv (K=4) + SiLU + grouped 16x16 gates (wave32
// shuffles) + per-CHUNK log-space reduction. The reference keeps only the
// last scan step, so cumsum/cumlogsumexp reduces to an associative
// (sum-of-log_f, logsumexp) pair, chunked over S for 8x more parallelism.
// grid = (INNER/256, NCHUNK, B); one lane per (b, chunk, c).
// ---------------------------------------------------------------------------
__global__ __launch_bounds__(256)
void conv_scan_chunk_kernel(const float* __restrict__ xin,    // (B,S,2*INNER)
                            const float* __restrict__ conv_w, // (INNER,1,4)
                            const float* __restrict__ conv_b,
                            const float* __restrict__ w_f,    // (G,16,16)
                            const float* __restrict__ w_i,
                            const float* __restrict__ w_h,
                            float* __restrict__ act,          // (B,S,INNER)
                            float* __restrict__ chA,          // (B,NCHUNK,INNER)
                            float* __restrict__ chL)          // (B,NCHUNK,INNER)
{
    const int c     = blockIdx.x * blockDim.x + threadIdx.x;  // channel
    const int chunk = blockIdx.y;
    const int b     = blockIdx.z;
    const int s0    = chunk * CHUNK;
    const int g     = c >> 4;
    const int o     = c & 15;
    const int baseLane = (threadIdx.x & 31) & 16;             // group base in wave

    const float cw0 = conv_w[c * KSZ + 0];
    const float cw1 = conv_w[c * KSZ + 1];
    const float cw2 = conv_w[c * KSZ + 2];
    const float cw3 = conv_w[c * KSZ + 3];
    const float cb  = conv_b[c];

    float wf[GSZ], wi[GSZ], wh[GSZ];
#pragma unroll
    for (int i = 0; i < GSZ; ++i) {
        const int base = (g * GSZ + o) * GSZ + i;
        wf[i] = w_f[base];
        wi[i] = w_i[base];
        wh[i] = w_h[base];
    }

    const float* xp = xin + (size_t)b * SLEN * (2 * INNERV) + c;
    float*       ap = act + (size_t)b * SLEN * INNERV + c;

    // conv history: read the 3 samples before the chunk (zeros for chunk 0)
    float x1 = 0.f, x2 = 0.f, x3 = 0.f;
    if (s0 > 0) {
        x1 = xp[(size_t)(s0 - 1) * (2 * INNERV)];
        x2 = xp[(size_t)(s0 - 2) * (2 * INNERV)];
        x3 = xp[(size_t)(s0 - 3) * (2 * INNERV)];
    }

    float Acc = 0.f;              // chunk-local cumsum(log_f + EPS)
    float mx  = -3.0e38f;         // running max for chunk logsumexp
    float se  = 0.f;              // running scaled sum

    for (int s = s0; s < s0 + CHUNK; ++s) {
        const float x0 = xp[(size_t)s * (2 * INNERV)];
        const float xc = cw0 * x3 + cw1 * x2 + cw2 * x1 + cw3 * x0 + cb;
        x3 = x2; x2 = x1; x1 = x0;

        const float a = silu_f(xc);
        ap[(size_t)s * INNERV] = a;

        // grouped 16x16 matvecs: gather group-mates' activations via shuffle
        float f = 0.f, ig = 0.f, hh = 0.f;
#pragma unroll
        for (int i = 0; i < GSZ; ++i) {
            const float av = __shfl(a, baseLane + i, 32);
            f  = fmaf(wf[i], av, f);
            ig = fmaf(wi[i], av, ig);
            hh = fmaf(wh[i], av, hh);
        }

        const float diff  = softplus_f(-f) - softplus_f(-ig);
        const float log_f = -softplus_f(diff);
        const float log_i = -softplus_f(-diff);
        const float lth   = (hh >= 0.f) ? __logf(hh + 0.5f + EPSV)
                                        : (-softplus_f(-hh) + EPSV);

        Acc += log_f + EPSV;
        const float t = (log_i + lth) - Acc;  // v_s - local a_star_s
        if (t > mx) { se = se * __expf(mx - t) + 1.f; mx = t; }
        else        { se += __expf(t - mx); }
    }

    const size_t oidx = ((size_t)b * NCHUNK + chunk) * INNERV + c;
    chA[oidx] = Acc;
    chL[oidx] = mx + __logf(se);
}

// ---------------------------------------------------------------------------
// Pass 2: combine the NCHUNK per-chunk partials per (b, c):
//   u_k = L_k - O_k,  O_{k+1} = O_k + A_k;  h_last = exp(O_total + lse(u))
// ---------------------------------------------------------------------------
__global__ __launch_bounds__(256)
void scan_combine_kernel(const float* __restrict__ chA,
                         const float* __restrict__ chL,
                         float* __restrict__ hlast)           // (B,INNER)
{
    const int tid = blockIdx.x * blockDim.x + threadIdx.x;    // 0..8191
    const int b   = tid / INNERV;
    const int c   = tid - b * INNERV;

    float O = 0.f, mx = -3.0e38f, se = 0.f;
#pragma unroll
    for (int k = 0; k < NCHUNK; ++k) {
        const size_t idx = ((size_t)b * NCHUNK + k) * INNERV + c;
        const float u = chL[idx] - O;
        if (u > mx) { se = se * __expf(mx - u) + 1.f; mx = u; }
        else        { se += __expf(u - mx); }
        O += chA[idx];
    }
    hlast[tid] = __expf(O + mx + __logf(se));
}

// ---------------------------------------------------------------------------
// Launch
// ---------------------------------------------------------------------------
extern "C" void kernel_launch(void* const* d_in, const int* in_sizes, int n_in,
                              void* d_out, int out_size, void* d_ws, size_t ws_size,
                              hipStream_t stream)
{
    (void)in_sizes; (void)n_in; (void)out_size; (void)ws_size;

    const float* x      = (const float*)d_in[0];
    const float* w_up   = (const float*)d_in[1];
    const float* b_up   = (const float*)d_in[2];
    const float* conv_w = (const float*)d_in[3];
    const float* conv_b = (const float*)d_in[4];
    const float* skip   = (const float*)d_in[5];
    const float* w_f    = (const float*)d_in[6];
    const float* w_i    = (const float*)d_in[7];
    const float* w_h    = (const float*)d_in[8];
    const float* w_down = (const float*)d_in[9];
    const float* b_down = (const float*)d_in[10];
    float* out = (float*)d_out;

    const int M = BATCH * SLEN;                      // 16384 rows

    // workspace layout (fp32): x_inner | act | h_last | chunk partials
    float* xin   = (float*)d_ws;                     // M * 4096
    float* actb  = xin   + (size_t)M * (2 * INNERV); // M * 2048
    float* hlast = actb  + (size_t)M * INNERV;       // 8192
    float* chA   = hlast + (size_t)BATCH * INNERV;   // B*NCHUNK*INNER = 65536
    float* chL   = chA   + (size_t)BATCH * NCHUNK * INNERV;

    dim3 blk(256);

    // 1) up-projection: xin = x @ w_up^T + b_up      (16384 x 4096 x 1024)
    gemm_bf16_kernel<0><<<dim3((2 * INNERV) / 128, M / 128), blk, 0, stream>>>(
        x, w_up, b_up, xin, M, 2 * INNERV, DIMV,
        nullptr, nullptr, nullptr);

    // 2a) conv + gates + chunked log-space reduction
    conv_scan_chunk_kernel<<<dim3(INNERV / 256, NCHUNK, BATCH), blk, 0, stream>>>(
        xin, conv_w, conv_b, w_f, w_i, w_h, actb, chA, chL);

    // 2b) combine chunk partials -> h_last
    scan_combine_kernel<<<dim3((BATCH * INNERV) / 256), blk, 0, stream>>>(
        chA, chL, hlast);

    // 3) fused epilogue + down-projection: out = h @ w_down^T + b_down
    //    h = (h_last + skip*act) * silu(z),  z = xin[..., INNER:]
    gemm_bf16_kernel<1><<<dim3(DIMV / 128, M / 128), blk, 0, stream>>>(
        actb, w_down, b_down, out, M, DIMV, INNERV,
        xin, skip, hlast);
}
